// MultiHeadLatentAttention_28269474742877
// MI455X (gfx1250) — compile-verified, matrix operand order checked by reading, not run on hardware
//
#include <hip/hip_runtime.h>
#include <hip/hip_bf16.h>

typedef __attribute__((ext_vector_type(16))) _Float16 v16h;
typedef __attribute__((ext_vector_type(8)))  float    v8f;
typedef __attribute__((ext_vector_type(4)))  unsigned u32x4;
typedef __attribute__((ext_vector_type(8)))  int      i32x8;
typedef __attribute__((ext_vector_type(4)))  int      i32x4;

// ---------------------------------------------------------------------------
// Problem constants
// ---------------------------------------------------------------------------
#define MB      2
#define SLEN    2048
#define DMODEL  2048
#define HEADS   16
#define NOPE_D  64
#define ROPE_D  32
#define V_D     64
#define QKD     96           // NOPE_D + ROPE_D
#define Q_RANK  768
#define KV_RANK 256
#define MROWS   (MB * SLEN)  // 4096

#define HAVE_TDM __has_builtin(__builtin_amdgcn_tensor_load_to_lds)

// ---------------------------------------------------------------------------
// WMMA fragment helpers (CDNA5 wave32, 16x16x32 f16 -> f32)
// A 16x32: lane = M (lane&15), k-halves: lanes 0-15 cover K 0-7/16-23,
//          lanes 16-31 cover K 8-15/24-31, 2 halves per VGPR along K.
// B 32x16: lane = N, same K packing.
// C/D 16x16 f32: lane = N, M = vgpr + 8*(lane>=16).
// ---------------------------------------------------------------------------
__device__ __forceinline__ int frag_koff(int v, int lane) {
  return (((lane >> 4) & 1) << 3) + ((v & 4) << 2) + ((v & 3) << 1);
}

__device__ __forceinline__ v16h load_frag_A(const _Float16* __restrict__ p,
                                            int ld, int m0, int k0, int lane) {
  const _Float16* row = p + (size_t)(m0 + (lane & 15)) * ld + k0;
  v16h a;
#pragma unroll
  for (int v = 0; v < 8; ++v) {
    int k = frag_koff(v, lane);
    a[2 * v]     = row[k];
    a[2 * v + 1] = row[k + 1];
  }
  return a;
}

__device__ __forceinline__ v16h load_frag_B(const _Float16* __restrict__ p,
                                            int ld, int k0, int n0, int lane) {
  const _Float16* col = p + (size_t)k0 * ld + n0 + (lane & 15);
  v16h b;
#pragma unroll
  for (int v = 0; v < 8; ++v) {
    int k = frag_koff(v, lane);
    b[2 * v]     = col[(size_t)k * ld];
    b[2 * v + 1] = col[(size_t)(k + 1) * ld];
  }
  return b;
}

__device__ __forceinline__ v8f wmma_f16(v16h a, v16h b, v8f c) {
  return __builtin_amdgcn_wmma_f32_16x16x32_f16(false, a, false, b,
                                                (short)0, c, false, false);
}

// ---------------------------------------------------------------------------
// Tensor Data Mover: DMA a 2-D f16 tile (tile_w x tile_h) from row-major
// global memory (row stride = stride_elems) into LDS at byte offset lds_off.
// Columns/rows beyond tensor_w/tensor_h are zero-filled by hardware.
// D# packing per CDNA5 ISA.
// ---------------------------------------------------------------------------
#if HAVE_TDM
__device__ __forceinline__ void tdm_load_2d(unsigned lds_off, const void* gptr,
                                            unsigned tile_w, unsigned tile_h,
                                            unsigned tensor_w, unsigned tensor_h,
                                            unsigned stride_elems) {
  unsigned long long ga = (unsigned long long)gptr;
  u32x4 g0;
  g0[0] = 1u;                                            // count=1, user mode
  g0[1] = lds_off;                                       // lds_addr (bytes)
  g0[2] = (unsigned)(ga & 0xffffffffull);                // global_addr[31:0]
  g0[3] = (unsigned)((ga >> 32) & 0x01ffffffull)         // global_addr[56:32]
          | (2u << 30);                                  // type=2 ("image")
  i32x8 g1;
  g1[0] = (int)(1u << 16);                               // data_size=1 (2B)
  g1[1] = (int)((tensor_w & 0xffffu) << 16);             // tensor_dim0[15:0]
  g1[2] = (int)(((tensor_w >> 16) & 0xffffu) |
                ((tensor_h & 0xffffu) << 16));           // dim0 hi | dim1 lo
  g1[3] = (int)(((tensor_h >> 16) & 0xffffu) |
                ((tile_w & 0xffffu) << 16));             // dim1 hi | tile_dim0
  g1[4] = (int)(tile_h & 0xffffu);                       // tile_dim1, tile_dim2=0
  g1[5] = (int)stride_elems;                             // dim0_stride[31:0]
  g1[6] = 0;                                             // dim0_stride hi
  g1[7] = 0;
  i32x4 gz = {0, 0, 0, 0};
#if defined(__clang_major__) && (__clang_major__ >= 23)
  i32x8 gz8 = {0, 0, 0, 0, 0, 0, 0, 0};
  __builtin_amdgcn_tensor_load_to_lds(g0, g1, gz, gz, gz8, 0);
#else
  __builtin_amdgcn_tensor_load_to_lds(g0, g1, gz, gz, 0);
#endif
}
#endif

__device__ __forceinline__ unsigned lds_offset_of(const void* p) {
  return (unsigned)(unsigned long long)p;  // generic LDS ptr: low 32 = offset
}

// ---------------------------------------------------------------------------
// Generic tiled GEMM: C(MxN) = A(MxK) @ B(KxN), A/B f16 row-major.
// Block = 256 threads = 8 waves; block tile 128x64; wave tile 16x64.
// A (128x32) and B (32x64) k-slabs DMA'd into LDS by the Tensor Data Mover
// (wave 0 issues, s_wait_tensorcnt, block barrier), fragments read from LDS.
// MODE 0: f32 row-major output.  MODE 1: f16 scattered to (B,H,S,headDim).
// ---------------------------------------------------------------------------
template <int MODE>
__global__ __launch_bounds__(256)
void gemm16(const _Float16* __restrict__ A, const _Float16* __restrict__ B,
            float* __restrict__ Cf, _Float16* __restrict__ Ch,
            int M, int N, int K, int headDim) {
  __shared__ _Float16 As[128][32];   // 8 KB
  __shared__ _Float16 Bs[32][64];    // 4 KB
  const int lane = threadIdx.x & 31;
  const int wave = threadIdx.x >> 5;
  const int m0blk = blockIdx.y * 128;
  const int n0 = blockIdx.x * 64;

  v8f acc[4];
#pragma unroll
  for (int t = 0; t < 4; ++t)
#pragma unroll
    for (int e = 0; e < 8; ++e) acc[t][e] = 0.0f;

  for (int k0 = 0; k0 < K; k0 += 32) {
    __syncthreads();  // previous iteration's LDS reads complete
#if HAVE_TDM
    if (wave == 0) {
      tdm_load_2d(lds_offset_of(&As[0][0]), A + (size_t)m0blk * K + k0,
                  /*tile*/ 32, 128, /*tensor*/ (unsigned)(K - k0),
                  (unsigned)(M - m0blk), (unsigned)K);
      tdm_load_2d(lds_offset_of(&Bs[0][0]), B + (size_t)k0 * N + n0,
                  /*tile*/ 64, 32, /*tensor*/ (unsigned)(N - n0),
                  (unsigned)(K - k0), (unsigned)N);
      __builtin_amdgcn_s_wait_tensorcnt((short)0);
    }
#else
    {
      // Manual staging fallback: 256 threads x 16 halves = A tile
      int idx = threadIdx.x * 16;
      int ar = idx >> 5, ac = idx & 31;
#pragma unroll
      for (int j = 0; j < 16; ++j)
        As[ar][ac + j] = A[(size_t)(m0blk + ar) * K + k0 + ac + j];
      // 256 threads x 8 halves = B tile
      int bi = threadIdx.x * 8;
      int kr = bi >> 6, nc = bi & 63;
#pragma unroll
      for (int j = 0; j < 8; ++j) {
        int n = n0 + nc + j;
        Bs[kr][nc + j] = (n < N) ? B[(size_t)(k0 + kr) * N + n] : (_Float16)0.0f;
      }
    }
#endif
    __syncthreads();
    v16h a = load_frag_A(&As[0][0], 32, wave * 16, 0, lane);
#pragma unroll
    for (int t = 0; t < 4; ++t) {
      if (n0 + t * 16 < N) {
        v16h b = load_frag_B(&Bs[0][0], 64, 0, t * 16, lane);
        acc[t] = wmma_f16(a, b, acc[t]);
      }
    }
  }

#pragma unroll
  for (int t = 0; t < 4; ++t) {
    int nb = n0 + t * 16;
    if (nb >= N) continue;
    int n = nb + (lane & 15);
#pragma unroll
    for (int v = 0; v < 8; ++v) {
      int m = m0blk + wave * 16 + v + (((lane >> 4) & 1) << 3);
      float val = acc[t][v];
      if (MODE == 0) {
        Cf[(size_t)m * N + n] = val;
      } else {
        int bb = m >> 11, s = m & (SLEN - 1);
        int h = n >> 6, d = n & 63;
        Ch[((size_t)(bb * HEADS + h) * SLEN + s) * headDim + d] = (_Float16)val;
      }
    }
  }
}

// ---------------------------------------------------------------------------
// RMSNorm: one block per row, output f16.
// ---------------------------------------------------------------------------
__global__ __launch_bounds__(256)
void rmsnorm16(const float* __restrict__ X, const float* __restrict__ g,
               _Float16* __restrict__ Y, int N) {
  __shared__ float red[9];
  const int r = blockIdx.x;
  const float* row = X + (size_t)r * N;
  float ss = 0.f;
  for (int i = threadIdx.x; i < N; i += 256) { float v = row[i]; ss += v * v; }
#pragma unroll
  for (int off = 16; off >= 1; off >>= 1) ss += __shfl_xor(ss, off, 32);
  if ((threadIdx.x & 31) == 0) red[threadIdx.x >> 5] = ss;
  __syncthreads();
  if (threadIdx.x == 0) {
    float t = 0.f;
    for (int i = 0; i < 8; ++i) t += red[i];
    red[8] = rsqrtf(t / (float)N + 1e-6f);
  }
  __syncthreads();
  float sc = red[8];
  for (int i = threadIdx.x; i < N; i += 256)
    Y[(size_t)r * N + i] = (_Float16)(row[i] * sc * g[i]);
}

// ---------------------------------------------------------------------------
// Elementwise f32 -> f16
// ---------------------------------------------------------------------------
__global__ __launch_bounds__(256)
void cvt_f32_f16(const float* __restrict__ s, _Float16* __restrict__ d, int n) {
  int i = blockIdx.x * 256 + threadIdx.x;
  if (i < n) d[i] = (_Float16)s[i];
}

// ---------------------------------------------------------------------------
// RoPE (reference uses sin table == cos table).
// ---------------------------------------------------------------------------
__global__ __launch_bounds__(256)
void rope_q(const float* __restrict__ raw, _Float16* __restrict__ Q) {
  int idx = blockIdx.x * 256 + threadIdx.x;       // MROWS * H * 16 pairs
  if (idx >= MROWS * HEADS * 16) return;
  int r = idx >> 8;
  int pr = idx & 255;
  int h = pr >> 4, i = pr & 15;
  int b = r >> 11, s = r & (SLEN - 1);
  const float* p = raw + (size_t)r * (HEADS * ROPE_D) + h * ROPE_D + 2 * i;
  float xr = p[0], xi = p[1];
  float c = __cosf((float)s * __powf(10000.f, -(float)(2 * i) / (float)ROPE_D));
  size_t base = ((size_t)(b * HEADS + h) * SLEN + s) * QKD + NOPE_D + 2 * i;
  Q[base]     = (_Float16)(xr * c - xi * c);
  Q[base + 1] = (_Float16)(xr * c + xi * c);
}

__global__ __launch_bounds__(256)
void rope_k(const float* __restrict__ raw, _Float16* __restrict__ Kb) {
  int idx = blockIdx.x * 256 + threadIdx.x;       // MROWS * 16 pairs
  if (idx >= MROWS * 16) return;
  int r = idx >> 4, i = idx & 15;
  int b = r >> 11, s = r & (SLEN - 1);
  float xr = raw[(size_t)r * ROPE_D + 2 * i] * (1.0f / HEADS);
  float xi = raw[(size_t)r * ROPE_D + 2 * i + 1] * (1.0f / HEADS);
  float c = __cosf((float)s * __powf(10000.f, -(float)(2 * i) / (float)ROPE_D));
  _Float16 o0 = (_Float16)(xr * c - xi * c);
  _Float16 o1 = (_Float16)(xr * c + xi * c);
#pragma unroll
  for (int h = 0; h < HEADS; ++h) {
    size_t base = ((size_t)(b * HEADS + h) * SLEN + s) * QKD + NOPE_D + 2 * i;
    Kb[base] = o0;
    Kb[base + 1] = o1;
  }
}

// ---------------------------------------------------------------------------
// Causal flash attention.  Block = 128 threads (4 waves); each wave owns 16
// query rows of a 64-row tile.  Per 32-key step: 6 WMMA for Q.K^T, streaming
// softmax with shfl_xor row reductions, P->LDS->A-frag, 4 WMMA for P.V.
// No block barriers (waves have divergent trip counts); wave-internal LDS
// ordering enforced with s_wait_dscnt.  Next-tile K/V prefetched.
// ---------------------------------------------------------------------------
__global__ __launch_bounds__(128)
void mla_attn(const _Float16* __restrict__ Q, const _Float16* __restrict__ Kb,
              const _Float16* __restrict__ V, _Float16* __restrict__ O) {
  __shared__ _Float16 Ps[4][16][40];
  const int lane = threadIdx.x & 31;
  const int wave = threadIdx.x >> 5;
  const int tile = blockIdx.x & 31;               // S/64 tiles
  const int h = (blockIdx.x >> 5) & (HEADS - 1);
  const int b = blockIdx.x >> 9;
  const _Float16* Qh = Q  + (size_t)(b * HEADS + h) * SLEN * QKD;
  const _Float16* Kh = Kb + (size_t)(b * HEADS + h) * SLEN * QKD;
  const _Float16* Vh = V  + (size_t)(b * HEADS + h) * SLEN * V_D;
  const int q0 = tile * 64 + wave * 16;
  const int hi8 = ((lane >> 4) & 1) << 3;
  const float scale = 0.10206207262f;             // 1/sqrt(96)

  v16h qf[3];
#pragma unroll
  for (int kk = 0; kk < 3; ++kk) qf[kk] = load_frag_A(Qh, QKD, q0, kk * 32, lane);

  v8f o[4];
  float rmax[8], rsum[8];
#pragma unroll
  for (int v = 0; v < 8; ++v) { rmax[v] = -INFINITY; rsum[v] = 0.f; }
#pragma unroll
  for (int t = 0; t < 4; ++t)
#pragma unroll
    for (int e = 0; e < 8; ++e) o[t][e] = 0.f;

  for (int k0 = 0; k0 <= q0 + 15; k0 += 32) {
    // Latency hints for the next key tile (global_prefetch_b8).
    __builtin_prefetch((const void*)(Kh + (size_t)(k0 + 32) * QKD + lane * 4), 0, 1);
    __builtin_prefetch((const void*)(Vh + (size_t)(k0 + 32) * V_D + lane * 4), 0, 1);

    v8f s0, s1;
#pragma unroll
    for (int e = 0; e < 8; ++e) { s0[e] = 0.f; s1[e] = 0.f; }
#pragma unroll
    for (int kk = 0; kk < 3; ++kk) {
      v16h b0 = load_frag_A(Kh, QKD, k0,      kk * 32, lane);  // K^T frag
      v16h b1 = load_frag_A(Kh, QKD, k0 + 16, kk * 32, lane);
      s0 = wmma_f16(qf[kk], b0, s0);
      s1 = wmma_f16(qf[kk], b1, s1);
    }
#pragma unroll
    for (int v = 0; v < 8; ++v) {
      int m = q0 + v + hi8;
      int n = k0 + (lane & 15);
      float x0 = (n      <= m) ? s0[v] * scale : -INFINITY;
      float x1 = (n + 16 <= m) ? s1[v] * scale : -INFINITY;
      float mx = fmaxf(x0, x1);
      mx = fmaxf(mx, __shfl_xor(mx, 1, 32));
      mx = fmaxf(mx, __shfl_xor(mx, 2, 32));
      mx = fmaxf(mx, __shfl_xor(mx, 4, 32));
      mx = fmaxf(mx, __shfl_xor(mx, 8, 32));
      float nm = fmaxf(rmax[v], mx);
      float alpha = __expf(rmax[v] - nm);
      rmax[v] = nm;
      float p0 = __expf(x0 - nm);
      float p1 = __expf(x1 - nm);
      float ps = p0 + p1;
      ps += __shfl_xor(ps, 1, 32);
      ps += __shfl_xor(ps, 2, 32);
      ps += __shfl_xor(ps, 4, 32);
      ps += __shfl_xor(ps, 8, 32);
      rsum[v] = rsum[v] * alpha + ps;
#pragma unroll
      for (int t = 0; t < 4; ++t) o[t][v] *= alpha;
      int mr = v + hi8;
      Ps[wave][mr][lane & 15]        = (_Float16)p0;
      Ps[wave][mr][(lane & 15) + 16] = (_Float16)p1;
    }
    asm volatile("s_wait_dscnt 0" ::: "memory");
    v16h aP = load_frag_A(&Ps[wave][0][0], 40, 0, 0, lane);
#pragma unroll
    for (int t = 0; t < 4; ++t) {
      v16h bv = load_frag_B(Vh + (size_t)k0 * V_D, V_D, 0, t * 16, lane);
      o[t] = wmma_f16(aP, bv, o[t]);
    }
  }

#pragma unroll
  for (int t = 0; t < 4; ++t)
#pragma unroll
    for (int v = 0; v < 8; ++v) {
      int m = v + hi8;
      int s = q0 + m;
      float val = o[t][v] / rsum[v];
      O[((size_t)b * SLEN + s) * (HEADS * V_D) + h * V_D + t * 16 + (lane & 15)] =
          (_Float16)val;
    }
}

// ---------------------------------------------------------------------------
// Host launcher
// ---------------------------------------------------------------------------
extern "C" void kernel_launch(void* const* d_in, const int* in_sizes, int n_in,
                              void* d_out, int out_size, void* d_ws, size_t ws_size,
                              hipStream_t stream) {
  (void)in_sizes; (void)n_in; (void)out_size; (void)ws_size;
  const float* x      = (const float*)d_in[0];
  const float* w_cq   = (const float*)d_in[1];
  const float* w_qn   = (const float*)d_in[2];
  const float* w_qr   = (const float*)d_in[3];
  const float* q_g    = (const float*)d_in[4];
  const float* w_ckv  = (const float*)d_in[5];
  const float* w_kn   = (const float*)d_in[6];
  const float* w_v    = (const float*)d_in[7];
  const float* kv_g   = (const float*)d_in[8];
  const float* w_kr   = (const float*)d_in[9];
  const float* w_pr   = (const float*)d_in[10];
  float* out = (float*)d_out;

  char* ws = (char*)d_ws;
  size_t off = 0;
  auto alloc = [&](size_t bytes) -> char* {
    char* p = ws + off;
    off = (off + bytes + 255) & ~(size_t)255;
    return p;
  };

  _Float16* xh      = (_Float16*)alloc((size_t)MROWS * DMODEL * 2);
  _Float16* wcq_h   = (_Float16*)alloc((size_t)DMODEL * Q_RANK * 2);
  _Float16* wqn_h   = (_Float16*)alloc((size_t)Q_RANK * HEADS * NOPE_D * 2);
  _Float16* wqr_h   = (_Float16*)alloc((size_t)Q_RANK * HEADS * ROPE_D * 2);
  _Float16* wckv_h  = (_Float16*)alloc((size_t)DMODEL * KV_RANK * 2);
  _Float16* wkn_h   = (_Float16*)alloc((size_t)KV_RANK * HEADS * NOPE_D * 2);
  _Float16* wv_h    = (_Float16*)alloc((size_t)KV_RANK * HEADS * V_D * 2);
  _Float16* wkr_h   = (_Float16*)alloc((size_t)DMODEL * ROPE_D * 2);
  _Float16* wpr_h   = (_Float16*)alloc((size_t)HEADS * V_D * DMODEL * 2);
  _Float16* normq_h = (_Float16*)alloc((size_t)MROWS * Q_RANK * 2);
  _Float16* normkv_h= (_Float16*)alloc((size_t)MROWS * KV_RANK * 2);
  _Float16* Qbuf    = (_Float16*)alloc((size_t)MB * HEADS * SLEN * QKD * 2);
  _Float16* Kbuf    = (_Float16*)alloc((size_t)MB * HEADS * SLEN * QKD * 2);
  _Float16* Vbuf    = (_Float16*)alloc((size_t)MB * HEADS * SLEN * V_D * 2);
  _Float16* attno_h = (_Float16*)alloc((size_t)MROWS * HEADS * V_D * 2);
  float* cq_f   = (float*)alloc((size_t)MROWS * Q_RANK * 4);
  float* ckv_f  = (float*)alloc((size_t)MROWS * KV_RANK * 4);
  float* qr_raw = (float*)alloc((size_t)MROWS * HEADS * ROPE_D * 4);
  float* kr_raw = (float*)alloc((size_t)MROWS * ROPE_D * 4);

  auto cvt = [&](const float* s, _Float16* d, int n) {
    cvt_f32_f16<<<(n + 255) / 256, 256, 0, stream>>>(s, d, n);
  };
  cvt(x,     xh,     MROWS * DMODEL);
  cvt(w_cq,  wcq_h,  DMODEL * Q_RANK);
  cvt(w_qn,  wqn_h,  Q_RANK * HEADS * NOPE_D);
  cvt(w_qr,  wqr_h,  Q_RANK * HEADS * ROPE_D);
  cvt(w_ckv, wckv_h, DMODEL * KV_RANK);
  cvt(w_kn,  wkn_h,  KV_RANK * HEADS * NOPE_D);
  cvt(w_v,   wv_h,   KV_RANK * HEADS * V_D);
  cvt(w_kr,  wkr_h,  DMODEL * ROPE_D);
  cvt(w_pr,  wpr_h,  HEADS * V_D * DMODEL);

  auto gemm_f32 = [&](const _Float16* A, const _Float16* B, float* C,
                      int M, int N, int K) {
    dim3 grid((N + 63) / 64, M / 128);
    gemm16<0><<<grid, 256, 0, stream>>>(A, B, C, nullptr, M, N, K, 0);
  };
  auto gemm_heads = [&](const _Float16* A, const _Float16* B, _Float16* C,
                        int M, int N, int K, int headDim) {
    dim3 grid((N + 63) / 64, M / 128);
    gemm16<1><<<grid, 256, 0, stream>>>(A, B, nullptr, C, M, N, K, headDim);
  };

  // Stage 1: latent projections from x
  gemm_f32(xh, wcq_h,  cq_f,   MROWS, Q_RANK, DMODEL);
  gemm_f32(xh, wckv_h, ckv_f,  MROWS, KV_RANK, DMODEL);
  gemm_f32(xh, wkr_h,  kr_raw, MROWS, ROPE_D, DMODEL);

  // Stage 2: RMSNorm
  rmsnorm16<<<MROWS, 256, 0, stream>>>(cq_f,  q_g,  normq_h,  Q_RANK);
  rmsnorm16<<<MROWS, 256, 0, stream>>>(ckv_f, kv_g, normkv_h, KV_RANK);

  // Stage 3: per-head projections
  gemm_heads(normq_h,  wqn_h, Qbuf,   MROWS, HEADS * NOPE_D, Q_RANK, QKD);
  gemm_f32 (normq_h,  wqr_h, qr_raw, MROWS, HEADS * ROPE_D, Q_RANK);
  gemm_heads(normkv_h, wkn_h, Kbuf,   MROWS, HEADS * NOPE_D, KV_RANK, QKD);
  gemm_heads(normkv_h, wv_h,  Vbuf,   MROWS, HEADS * V_D,    KV_RANK, V_D);

  // Stage 4: RoPE into Q/K rope slots
  rope_q<<<(MROWS * HEADS * 16 + 255) / 256, 256, 0, stream>>>(qr_raw, Qbuf);
  rope_k<<<(MROWS * 16 + 255) / 256, 256, 0, stream>>>(kr_raw, Kbuf);

  // Stage 5: causal flash attention
  mla_attn<<<MB * HEADS * (SLEN / 64), 128, 0, stream>>>(Qbuf, Kbuf, Vbuf, attno_h);

  // Stage 6: output projection -> f32 out
  gemm_f32(attno_h, wpr_h, out, MROWS, DMODEL, HEADS * V_D);
}
